// CausalSelfAttention_88046829568441
// MI455X (gfx1250) — matrix operand order repaired and checked by reading, not compiled, not run on hardware
//
#include <hip/hip_runtime.h>

// ---------------------------------------------------------------------------
// Types for CDNA5 WMMA (wave32): v_wmma_f32_16x16x32_bf16
// ---------------------------------------------------------------------------
typedef __attribute__((ext_vector_type(16))) __bf16        bf16x16;
typedef __attribute__((ext_vector_type(8)))  float         f32x8;
typedef __attribute__((ext_vector_type(8)))  unsigned int  u32x8;

__device__ __forceinline__ unsigned short f2bf(float f) {
  unsigned int x = __float_as_uint(f);
  x += 0x7fffu + ((x >> 16) & 1u);          // round-to-nearest-even
  return (unsigned short)(x >> 16);
}

// A-matrix fragment (16x32 bf16, M x K). Storage: [m][k], k contiguous.
// Lane l: m = l&15, half = l>>4.
//  VGPR i (0..3):  K = 8*half + 2i, 2i+1
//  VGPR i (4..7):  K = 16 + 8*half + 2(i-4), +1
__device__ __forceinline__ bf16x16 frag_a(const unsigned short* base, int ld, int lane) {
  const int m = lane & 15, hh = lane >> 4;
  const unsigned short* p = base + m * ld;
  u32x8 u;
#pragma unroll
  for (int i = 0; i < 4; ++i)
    u[i] = *(const unsigned int*)(p + 8 * hh + 2 * i);
#pragma unroll
  for (int i = 0; i < 4; ++i)
    u[4 + i] = *(const unsigned int*)(p + 16 + 8 * hh + 2 * i);
  return __builtin_bit_cast(bf16x16, u);
}

// B-matrix fragment (32x16 bf16, K x N). Storage: [n][k] (i.e. W layout),
// k contiguous. Lane l: n = l&15, half = l>>4; VGPR i (0..3): K = 8*half + 2i, +1;
// VGPR i (4..7): K = 16 + 8*half + 2(i-4), +1.
__device__ __forceinline__ bf16x16 frag_b(const unsigned short* base, int ld, int lane) {
  const int n = lane & 15, hh = lane >> 4;
  const unsigned short* p = base + n * ld;
  u32x8 u;
#pragma unroll
  for (int i = 0; i < 8; ++i)
    u[i] = *(const unsigned int*)(p + (i >> 2) * 16 + 8 * hh + 2 * (i & 3));
  return __builtin_bit_cast(bf16x16, u);
}

// ---------------------------------------------------------------------------
// Generic batched GEMM:  C[z] = A[z] (MxK, f32) * W[z]^T (N x K, f32) + bias
// M multiple of 64, N multiple of 64, K multiple of 32.
// Block 256 threads (8 waves); 64x64 tile; wave -> 16x32 (2 wmma accums).
// ---------------------------------------------------------------------------
__global__ void __launch_bounds__(256) gemm_bf16(
    const float* __restrict__ A, int lda, int astride,
    const float* __restrict__ W, int ldw, int wstride,
    const float* __restrict__ bias, int bstride,
    float* __restrict__ C, int ldc, int cstride,
    int K) {
  __shared__ unsigned short As[64 * 32];
  __shared__ unsigned short Bs[64 * 32];
  const int tid = threadIdx.x, lane = tid & 31, wave = tid >> 5;
  const int wm = wave & 3, wn = wave >> 2;           // 4 m-tiles x 2 n-slabs
  const int z = blockIdx.z;
  const float* Ab = A + z * astride + (blockIdx.y * 64) * lda;
  const float* Wb = W + z * wstride + (blockIdx.x * 64) * ldw;
  f32x8 acc0, acc1;
#pragma unroll
  for (int v = 0; v < 8; ++v) { acc0[v] = 0.0f; acc1[v] = 0.0f; }

  const int lrow = tid >> 2, kseg = (tid & 3) * 8;
  for (int k0 = 0; k0 < K; k0 += 32) {
    const float* ap = Ab + lrow * lda + k0 + kseg;
    const float* wp = Wb + lrow * ldw + k0 + kseg;
#pragma unroll
    for (int i = 0; i < 8; ++i) As[lrow * 32 + kseg + i] = f2bf(ap[i]);
#pragma unroll
    for (int i = 0; i < 8; ++i) Bs[lrow * 32 + kseg + i] = f2bf(wp[i]);
    __syncthreads();
    bf16x16 a  = frag_a(&As[wm * 16 * 32], 32, lane);
    bf16x16 b0 = frag_b(&Bs[(wn * 32) * 32], 32, lane);
    bf16x16 b1 = frag_b(&Bs[(wn * 32 + 16) * 32], 32, lane);
    acc0 = __builtin_amdgcn_wmma_f32_16x16x32_bf16(false, a, false, b0, (short)0, acc0, false, false);
    acc1 = __builtin_amdgcn_wmma_f32_16x16x32_bf16(false, a, false, b1, (short)0, acc1, false, false);
    __syncthreads();
  }

  const int hh = lane >> 4, nn = lane & 15;
  const int col0 = blockIdx.x * 64 + wn * 32;
  float* Cb = C + z * cstride;
  float b0v = bias ? bias[z * bstride + col0 + nn] : 0.0f;
  float b1v = bias ? bias[z * bstride + col0 + 16 + nn] : 0.0f;
#pragma unroll
  for (int v = 0; v < 8; ++v) {
    int row = blockIdx.y * 64 + wm * 16 + v + 8 * hh;
    Cb[row * ldc + col0 + nn]      = acc0[v] + b0v;
    Cb[row * ldc + col0 + 16 + nn] = acc1[v] + b1v;
  }
}

// ---------------------------------------------------------------------------
// RoPE + head split: qkv (2048 x 3072) -> Q,K (rotated), V in (B,H,T,64)
// ---------------------------------------------------------------------------
__global__ void __launch_bounds__(256) rope_split(
    const float* __restrict__ qkv, const float* __restrict__ cosb,
    const float* __restrict__ sinb, float* __restrict__ Q,
    float* __restrict__ K, float* __restrict__ V) {
  int idx = blockIdx.x * 256 + threadIdx.x;     // B*H*T*32 = 1048576
  int i = idx & 31, t = (idx >> 5) & 1023, h = (idx >> 15) & 15, b = idx >> 19;
  const float* row = qkv + (b * 1024 + t) * 3072 + h * 64;
  float c = cosb[t * 32 + i], s = sinb[t * 32 + i];
  float qe = row[2 * i], qo = row[2 * i + 1];
  float ke = row[1024 + 2 * i], ko = row[1024 + 2 * i + 1];
  int o = ((b * 16 + h) * 1024 + t) * 64;
  Q[o + 2 * i] = qe * c - qo * s;  Q[o + 2 * i + 1] = qe * s + qo * c;
  K[o + 2 * i] = ke * c - ko * s;  K[o + 2 * i + 1] = ke * s + ko * c;
  V[o + 2 * i] = row[2048 + 2 * i];
  V[o + 2 * i + 1] = row[2048 + 2 * i + 1];
}

// ---------------------------------------------------------------------------
// Flash attention, template head dim DH in {16, 64}. 64-query block per WG.
// Streaming softmax in LDS; S = Q K^T and O += P V via bf16 WMMA.
// z decode: th = z % nTH; b = (z/nTH) % nB; h = z/(nTH*nB).
// ---------------------------------------------------------------------------
template <int DH>
__global__ void __launch_bounds__(256) flash_kernel(
    const float* __restrict__ Q0, const float* __restrict__ K0,
    const float* __restrict__ V0, float* __restrict__ O0,
    int ld, int ldo,
    int SH, int SB, int ST,
    int OSH, int OSB, int OST,
    int nB, int nTH, float scale) {
  constexpr int DPAD = (DH < 32) ? 32 : DH;   // pad K-dim of S-gemm to 32
  constexpr int CPT = DH / 4;                 // cols per thread on tile loads
  constexpr int NT = DH / 16;                 // O n-tiles
  constexpr int TILES = 4 * NT;               // total O tiles (64 x DH)
  constexpr int TPW = (TILES + 7) / 8;        // O tiles per wave

  __shared__ unsigned short Qs[64 * DPAD];
  __shared__ unsigned short Ks[64 * DPAD];
  __shared__ unsigned short Vts[DH * 64];     // [d][kv]  (B operand of P@V)
  __shared__ float Sbuf[64 * 64];
  __shared__ unsigned short Pbuf[64 * 64];
  __shared__ float mrun[64], lrun[64], corr[64];

  const int tid = threadIdx.x, lane = tid & 31, wave = tid >> 5;
  const int hh = lane >> 4, nn = lane & 15;
  const int z = blockIdx.z;
  const int th = z % nTH, rem = z / nTH, b = rem % nB, h = rem / nB;
  const float* Qg = Q0 + h * SH + b * SB + th * ST;
  const float* Kg = K0 + h * SH + b * SB + th * ST;
  const float* Vg = V0 + h * SH + b * SB + th * ST;
  float* Og = O0 + h * OSH + b * OSB + th * OST;
  const int q0 = blockIdx.x * 64;

  { // load Q block into LDS (bf16, padded)
    int r = tid >> 2, c0 = (tid & 3) * CPT;
    const float* qp = Qg + (q0 + r) * ld + c0;
#pragma unroll
    for (int i = 0; i < CPT; ++i) Qs[r * DPAD + c0 + i] = f2bf(qp[i]);
    if constexpr (DH < DPAD) {
      int p0 = DH + (tid & 3) * ((DPAD - DH) / 4);
#pragma unroll
      for (int i = 0; i < (DPAD - DH) / 4; ++i) Qs[r * DPAD + p0 + i] = 0;
    }
  }
  if (tid < 64) { mrun[tid] = -1e30f; lrun[tid] = 0.0f; }

  f32x8 oacc[TPW];
#pragma unroll
  for (int j = 0; j < TPW; ++j)
#pragma unroll
    for (int v = 0; v < 8; ++v) oacc[j][v] = 0.0f;

  const int ktmax = q0 >> 6;                  // causal: only kv-tiles <= q-tile
  for (int kt = 0; kt <= ktmax; ++kt) {
    __syncthreads();
    { // load K tile [n][d] and V tile transposed [d][kv]
      int r = tid >> 2, c0 = (tid & 3) * CPT;
      const float* kp = Kg + (kt * 64 + r) * ld + c0;
#pragma unroll
      for (int i = 0; i < CPT; ++i) Ks[r * DPAD + c0 + i] = f2bf(kp[i]);
      if constexpr (DH < DPAD) {
        int p0 = DH + (tid & 3) * ((DPAD - DH) / 4);
#pragma unroll
        for (int i = 0; i < (DPAD - DH) / 4; ++i) Ks[r * DPAD + p0 + i] = 0;
      }
      const float* vp = Vg + (kt * 64 + r) * ld + c0;
#pragma unroll
      for (int i = 0; i < CPT; ++i) Vts[(c0 + i) * 64 + r] = f2bf(vp[i]);
    }
    __syncthreads();
    { // S tiles: wave w -> sm = w>>1, sn = (w&1)*2 + {0,1}
      int sm = wave >> 1;
#pragma unroll
      for (int j = 0; j < 2; ++j) {
        int sn = (wave & 1) * 2 + j;
        f32x8 s;
#pragma unroll
        for (int v = 0; v < 8; ++v) s[v] = 0.0f;
#pragma unroll
        for (int ks = 0; ks < DPAD / 32; ++ks) {
          bf16x16 a = frag_a(Qs + sm * 16 * DPAD + ks * 32, DPAD, lane);
          bf16x16 bb = frag_b(Ks + sn * 16 * DPAD + ks * 32, DPAD, lane);
          s = __builtin_amdgcn_wmma_f32_16x16x32_bf16(false, a, false, bb, (short)0, s, false, false);
        }
#pragma unroll
        for (int v = 0; v < 8; ++v)
          Sbuf[(sm * 16 + v + 8 * hh) * 64 + sn * 16 + nn] = s[v] * scale;
      }
    }
    __syncthreads();
    if (tid < 64) { // online softmax row update (causal mask by index)
      int qg = q0 + tid;
      int climit = qg - kt * 64; if (climit > 63) climit = 63;
      float m_old = mrun[tid], rowmax = -1e30f;
      for (int c = 0; c <= climit; ++c) rowmax = fmaxf(rowmax, Sbuf[tid * 64 + c]);
      float m_new = fmaxf(m_old, rowmax);
      float cr = __expf(m_old - m_new);
      float sum = 0.0f;
      for (int c = 0; c < 64; ++c) {
        float p = (c <= climit) ? __expf(Sbuf[tid * 64 + c] - m_new) : 0.0f;
        Pbuf[tid * 64 + c] = f2bf(p);
        sum += p;
      }
      lrun[tid] = lrun[tid] * cr + sum;
      mrun[tid] = m_new;
      corr[tid] = cr;
    }
    __syncthreads();
    // O update: rescale then O += P @ V (kv = 64 -> two K-steps)
#pragma unroll
    for (int j = 0; j < TPW; ++j) {
      int ti = wave * TPW + j;
      if (ti < TILES) {
        int om = ti / NT, on = ti % NT;
#pragma unroll
        for (int v = 0; v < 8; ++v) oacc[j][v] *= corr[om * 16 + v + 8 * hh];
#pragma unroll
        for (int ks = 0; ks < 2; ++ks) {
          bf16x16 a = frag_a(Pbuf + om * 16 * 64 + ks * 32, 64, lane);
          bf16x16 bb = frag_b(Vts + on * 16 * 64 + ks * 32, 64, lane);
          oacc[j] = __builtin_amdgcn_wmma_f32_16x16x32_bf16(false, a, false, bb, (short)0, oacc[j], false, false);
        }
      }
    }
  }
  __syncthreads();
  if (tid < 64) corr[tid] = 1.0f / lrun[tid];
  __syncthreads();
#pragma unroll
  for (int j = 0; j < TPW; ++j) {
    int ti = wave * TPW + j;
    if (ti < TILES) {
      int om = ti / NT, on = ti % NT;
#pragma unroll
      for (int v = 0; v < 8; ++v) {
        int rl = om * 16 + v + 8 * hh;
        Og[(q0 + rl) * ldo + on * 16 + nn] = oacc[j][v] * corr[rl];
      }
    }
  }
}

// ---------------------------------------------------------------------------
// ctx = mean over T of hyper output (H, B*T, 64). One block per (h,b).
// ---------------------------------------------------------------------------
__global__ void __launch_bounds__(256) mean_kernel(const float* __restrict__ in,
                                                   float* __restrict__ ctx) {
  __shared__ float part[256];
  int hb = blockIdx.x;                         // h*2 + b
  const float* base = in + hb * 65536;
  int d = threadIdx.x & 63, seg = threadIdx.x >> 6;
  float s = 0.0f;
  for (int t = seg * 256; t < seg * 256 + 256; ++t) s += base[t * 64 + d];
  part[threadIdx.x] = s;
  __syncthreads();
  if (threadIdx.x < 64)
    ctx[hb * 64 + threadIdx.x] =
        (part[threadIdx.x] + part[64 + threadIdx.x] + part[128 + threadIdx.x] +
         part[192 + threadIdx.x]) * (1.0f / 1024.0f);
}

// ---------------------------------------------------------------------------
// A = ctx @ toA^T, Bm = ctx @ toB^T  (per head). 32768 output elements.
// ---------------------------------------------------------------------------
__global__ void __launch_bounds__(256) ab_kernel(
    const float* __restrict__ ctx, const float* __restrict__ toA,
    const float* __restrict__ toB, float* __restrict__ Am,
    float* __restrict__ Bm) {
  int idx = blockIdx.x * 256 + threadIdx.x;    // [which][h][b][n]
  int n = idx & 511, b = (idx >> 9) & 1, h = (idx >> 10) & 15, which = idx >> 14;
  const float* c = ctx + (h * 2 + b) * 64;
  const float* w = (which ? toB : toA) + h * 512 * 64 + n * 64;
  float s = 0.0f;
#pragma unroll
  for (int d = 0; d < 64; ++d) s += c[d] * w[d];
  (which ? Bm : Am)[(h * 2 + b) * 512 + n] = s;
}

// ---------------------------------------------------------------------------
// v += SCALE * (v @ A) @ Bm   (rank 8). One 64-thread block per (b,h,t) row.
// ---------------------------------------------------------------------------
__global__ void __launch_bounds__(64) delta_kernel(float* __restrict__ V,
                                                   const float* __restrict__ Am,
                                                   const float* __restrict__ Bm) {
  __shared__ float vrow[64];
  __shared__ float rr[8];
  int bid = blockIdx.x;
  int t = bid & 1023, h = (bid >> 10) & 15, b = bid >> 14;
  float* vp = V + ((b * 16 + h) * 1024 + t) * 64;
  int d = threadIdx.x;
  vrow[d] = vp[d];
  __syncthreads();
  if (d < 8) {
    const float* ap = Am + (h * 2 + b) * 512;   // [d][r], r stride 1
    float s = 0.0f;
    for (int dd = 0; dd < 64; ++dd) s += vrow[dd] * ap[dd * 8 + d];
    rr[d] = s;
  }
  __syncthreads();
  const float* bp = Bm + (h * 2 + b) * 512;     // [r][o]
  float delta = 0.0f;
#pragma unroll
  for (int j = 0; j < 8; ++j) delta += rr[j] * bp[j * 64 + d];
  vp[d] += 2.0f * delta;                        // SCALE = 16/8
}

// ---------------------------------------------------------------------------
extern "C" void kernel_launch(void* const* d_in, const int* in_sizes, int n_in,
                              void* d_out, int out_size, void* d_ws, size_t ws_size,
                              hipStream_t stream) {
  (void)in_sizes; (void)n_in; (void)out_size; (void)ws_size;
  const float* x        = (const float*)d_in[0];
  const float* cosb     = (const float*)d_in[1];
  const float* sinb     = (const float*)d_in[2];
  const float* c_attn_w = (const float*)d_in[3];
  const float* c_attn_b = (const float*)d_in[4];
  const float* c_proj_w = (const float*)d_in[5];
  const float* c_proj_b = (const float*)d_in[6];
  const float* hin_w    = (const float*)d_in[7];
  const float* hin_b    = (const float*)d_in[8];
  const float* hout_w   = (const float*)d_in[9];
  const float* hout_b   = (const float*)d_in[10];
  const float* toA_w    = (const float*)d_in[11];
  const float* toB_w    = (const float*)d_in[12];
  float* out = (float*)d_out;

  float* ws   = (float*)d_ws;
  float* QKV  = ws;                 // 6291456 floats (reused as tiny qkv)
  float* Qb   = QKV + 6291456;      // 2097152 each below
  float* Kb   = Qb + 2097152;
  float* Vb   = Kb + 2097152;
  float* OUTa = Vb + 2097152;
  float* OUTb = OUTa + 2097152;
  float* OT   = OUTb + 2097152;
  float* Y    = OT + 2097152;
  float* CTX  = Y + 2097152;        // 2048
  float* AM   = CTX + 2048;         // 16384
  float* BM   = AM + 16384;         // 16384

  dim3 blk(256);

  // 1) qkv = x @ c_attn_w^T + b          (2048 x 3072 x 1024)
  gemm_bf16<<<dim3(48, 32, 1), blk, 0, stream>>>(
      x, 1024, 0, c_attn_w, 1024, 0, c_attn_b, 0, QKV, 3072, 0, 1024);

  // 2) RoPE + split into (B,H,T,64)
  rope_split<<<dim3(4096), blk, 0, stream>>>(QKV, cosb, sinb, Qb, Kb, Vb);

  // 3) hyper path: 2 tiny-MHA layers, batched over H=16 heads (grid.z)
  float* TQ = QKV;                  // tiny qkv buffer (H, 2048, 192)
  for (int l = 0; l < 2; ++l) {
    const float* Ain = (l == 0) ? x : OUTa;
    int lda  = (l == 0) ? 1024 : 64;
    int astr = (l == 0) ? 64 : 131072;      // l=0: per-head column slice of x
    gemm_bf16<<<dim3(3, 32, 16), blk, 0, stream>>>(
        Ain, lda, astr,
        hin_w + l * 192 * 64, 64, 2 * 192 * 64,
        hin_b + l * 192, 2 * 192,
        TQ, 192, 393216, 64);
    flash_kernel<16><<<dim3(16, 1, 128), blk, 0, stream>>>(
        TQ, TQ + 64, TQ + 128, OT,
        192, 64,
        393216, 196608, 16,        // q/k/v strides per (h, b, tiny-head)
        131072, 65536, 16,         // output strides
        2, 4, 0.25f);
    float* Odst = (l == 0) ? OUTa : OUTb;
    gemm_bf16<<<dim3(1, 32, 16), blk, 0, stream>>>(
        OT, 64, 131072,
        hout_w + l * 64 * 64, 64, 2 * 64 * 64,
        hout_b + l * 64, 2 * 64,
        Odst, 64, 131072, 64);
  }

  // 4) ctx = mean over T; A/Bm low-rank generators
  mean_kernel<<<dim3(32), blk, 0, stream>>>(OUTb, CTX);
  ab_kernel<<<dim3(128), blk, 0, stream>>>(CTX, toA_w, toB_w, AM, BM);

  // 5) v += SCALE * (v A) Bm
  delta_kernel<<<dim3(32768), dim3(64), 0, stream>>>(Vb, AM, BM);

  // 6) main attention -> Y in (B, T, H*64) layout
  flash_kernel<64><<<dim3(16, 1, 32), blk, 0, stream>>>(
      Qb, Kb, Vb, Y,
      64, 1024,
      1048576, 65536, 0,           // decoded: "h"->batch, "b"->head
      1048576, 64, 0,
      16, 1, 0.125f);

  // 7) out = Y @ c_proj_w^T + b
  gemm_bf16<<<dim3(16, 32, 1), blk, 0, stream>>>(
      Y, 1024, 0, c_proj_w, 1024, 0, c_proj_b, 0, out, 1024, 0, 1024);
}
